// UnifiedInteractionLayer_30777735643334
// MI455X (gfx1250) — matrix operand order repaired and checked by reading, not compiled, not run on hardware
//
#include <hip/hip_runtime.h>
#include <hip/hip_bf16.h>
#include <math.h>

// ---------------- constants ----------------
constexpr int N_NODES = 10000;
constexpr int N_EDGES = 160000;
constexpr int S = 128, V = 64, L = 32, T = 128;
constexpr int F = S + 6 * V + 5 * L;          // 672
constexpr int KA = 448;                        // EDGE_DIM=444 padded to mult of 32
constexpr int EDGE_DIM = 444;
constexpr int NW = 832;
constexpr float DEG_NORM = 0.25f;              // 1/sqrt(160000/10000)
constexpr float EPS = 1e-6f;

// w slice offsets inside NW=832
// w0a[0:128] w0b[128:192] w0c[192:224] w1a[224:352] w1b[352:416] w1c[416:480]
// w2a[480:544] w2b[544:608] w3a[608:640] w3b[640:768] w3c[768:832]

// ---------------- workspace layout (bytes) ----------------
constexpr size_t ESC_OFF   = 0;                                   // E*448 bf16
constexpr size_t GEOM_OFF  = ESC_OFF + (size_t)N_EDGES * KA * 2;  // E*8 f32 (Y1[3],Y2[5])
constexpr size_t WBF_OFF   = GEOM_OFF + (size_t)N_EDGES * 8 * 4;  // E*832 bf16
constexpr size_t MSG_OFF   = WBF_OFF + (size_t)N_EDGES * NW * 2;  // N*672 f32
constexpr size_t ADA_OFF   = MSG_OFF + (size_t)N_NODES * F * 4;   // N*416 f32
constexpr size_t TPWT_OFF  = ADA_OFF + (size_t)N_NODES * 416 * 4; // 832*448 bf16
constexpr size_t PROJS_OFF = TPWT_OFF + (size_t)NW * KA * 2;      // 128*128 bf16
constexpr size_t P1O_OFF   = PROJS_OFF + 128 * 128 * 2;           // 64*64 bf16
constexpr size_t P1E_OFF   = P1O_OFF + 64 * 64 * 2;               // 64*64 bf16
constexpr size_t P2T_OFF   = P1E_OFF + 64 * 64 * 2;               // 32*32 bf16
constexpr size_t ADAT_OFF  = P2T_OFF + 32 * 32 * 2;               // 416*128 bf16

// ---------------- WMMA plumbing ----------------
typedef __attribute__((ext_vector_type(16))) __bf16 v16bf;
typedef __attribute__((ext_vector_type(8)))  float  v8f;

__device__ __forceinline__ v8f wmma_bf16(v16bf a, v16bf b, v8f c) {
  return __builtin_amdgcn_wmma_f32_16x16x32_bf16(false, a, false, b, (short)0, c,
                                                 false, false);
}

// A fragment (16x32 bf16, row-major source [M][ldk]).
// Lane L (L<16): row M=L, K = k0+{0..7} (elems 0..7) and k0+16+{0..7} (elems 8..15)
// Lane L+16:     row M=L, K = k0+8+{0..7}               and k0+24+{0..7}
__device__ __forceinline__ v16bf load_a_bf16(const __bf16* base, int ldk, int row0,
                                             int k0, int lane) {
  int m = lane & 15, hi = (lane >> 4) & 1;
  const __bf16* p = base + (size_t)(row0 + m) * ldk + k0 + hi * 8;
  v16bf r;
#pragma unroll
  for (int i = 0; i < 8; ++i) { r[i] = p[i]; r[i + 8] = p[16 + i]; }
  return r;
}

// A fragment from f32 source, converting, with arbitrary k stride.
// row_base points at (row of this lane) with fixed offsets folded in.
__device__ __forceinline__ v16bf load_a_f32(const float* row_base, int kstride,
                                            int k0, int hi) {
  v16bf r;
#pragma unroll
  for (int i = 0; i < 8; ++i) {
    r[i]     = (__bf16)row_base[(size_t)(k0 + hi * 8 + i) * kstride];
    r[i + 8] = (__bf16)row_base[(size_t)(k0 + 16 + hi * 8 + i) * kstride];
  }
  return r;
}

// B fragment (32x16 bf16) from transposed source stored [N][ldk] row-major.
// Lane L (<16): col N=L, K = k0+{0..7},k0+16+{0..7}; lane L+16: col N=L, K = k0+8+{0..7},k0+24+{0..7}
__device__ __forceinline__ v16bf load_b_bf16(const __bf16* baseT, int ldk, int n0,
                                             int k0, int lane) {
  int n = lane & 15, hi = (lane >> 4) & 1;
  const __bf16* p = baseT + (size_t)(n0 + n) * ldk + k0 + hi * 8;
  v16bf r;
#pragma unroll
  for (int i = 0; i < 8; ++i) { r[i] = p[i]; r[i + 8] = p[16 + i]; }
  return r;
}

// ---------------- K0: weight prep (transpose + f32->bf16) ----------------
__global__ void k0_prep(const float* tpW, const float* projs, const float* p1o,
                        const float* p1e, const float* p2, const float* adaW,
                        __bf16* tpWt, __bf16* projsT, __bf16* p1oT, __bf16* p1eT,
                        __bf16* p2T, __bf16* adaT) {
  const int T0 = NW * KA, T1 = 128 * 128, T2 = 64 * 64, T3 = 64 * 64, T4 = 32 * 32,
            T5 = 416 * 128;
  int total = T0 + T1 + T2 + T3 + T4 + T5;
  for (int i = blockIdx.x * blockDim.x + threadIdx.x; i < total;
       i += gridDim.x * blockDim.x) {
    int j = i;
    if (j < T0) { int n = j / KA, k = j % KA;
      tpWt[j] = (__bf16)(k < EDGE_DIM ? tpW[(size_t)k * NW + n] : 0.f); continue; }
    j -= T0;
    if (j < T1) { int n = j / 128, k = j % 128;
      projsT[j] = (__bf16)projs[k * 128 + n]; continue; }
    j -= T1;
    if (j < T2) { int n = j / 64, k = j % 64;
      p1oT[j] = (__bf16)p1o[k * 64 + n]; continue; }
    j -= T2;
    if (j < T3) { int n = j / 64, k = j % 64;
      p1eT[j] = (__bf16)p1e[k * 64 + n]; continue; }
    j -= T3;
    if (j < T4) { int n = j / 32, k = j % 32;
      p2T[j] = (__bf16)p2[k * 32 + n]; continue; }
    j -= T4;
    { int n = j / 128, k = j % 128; adaT[j] = (__bf16)adaW[(size_t)k * 416 + n]; }
  }
}

// ---------------- K1: edge scalar build (wave per edge) ----------------
__global__ __launch_bounds__(256) void k1_edge_prep(
    const float* h, const float* coords, const int* eidx, const int* etype,
    const int* ebt, const int* ebc, const int* ebr, const int* ebs,
    const float* eref, const float* temb, const float* type_emb,
    const float* bt_emb, const float* conj_emb, const float* ring_emb,
    const float* stereo_emb, const float* refW, const float* refb, __bf16* esc,
    float* geom) {
  int e = (blockIdx.x * blockDim.x + threadIdx.x) >> 5;
  int lane = threadIdx.x & 31;
  if (e >= N_EDGES) return;
  int src = eidx[e], dst = eidx[N_EDGES + e];
  float dx = coords[dst * 3 + 0] - coords[src * 3 + 0];
  float dy = coords[dst * 3 + 1] - coords[src * 3 + 1];
  float dz = coords[dst * 3 + 2] - coords[src * 3 + 2];
  float dist = sqrtf(dx * dx + dy * dy + dz * dz + 1e-12f);
  float inv = 1.f / dist;
  float ux = dx * inv, uy = dy * inv, uz = dz * inv;
  const float SQ3 = 1.73205081f, S15 = 3.87298335f, HS5 = 1.11803399f,
              HS15 = 1.93649167f;
  float y1x = SQ3 * ux, y1y = SQ3 * uy, y1z = SQ3 * uz;
  float y2a = S15 * ux * uy, y2b = S15 * uy * uz, y2c = HS5 * (3.f * uz * uz - 1.f),
        y2d = S15 * ux * uz, y2e = HS15 * (ux * ux - uy * uy);
  if (lane == 0) {
    float* g = geom + (size_t)e * 8;
    g[0] = y1x; g[1] = y1y; g[2] = y1z;
    g[3] = y2a; g[4] = y2b; g[5] = y2c; g[6] = y2d; g[7] = y2e;
  }
  float ref = eref[e];
  float has = ref > 0.f ? 1.f : 0.f;
  float dd = dist - ref;
  float add = fabsf(dd);
  int ety = etype[e], bty = ebt[e], bcj = ebc[e], brg = ebr[e], bst = ebs[e];
  __bf16* out = esc + (size_t)e * KA;
  const float W_RBF = 10.f / 15.f;
  for (int c = lane; c < KA; c += 32) {
    float v;
    if (c < 16) {
      float t = (dist - c * W_RBF) / W_RBF;
      v = expf(-t * t);
    } else if (c < 32) v = type_emb[ety * 16 + (c - 16)];
    else if (c < 40) v = bt_emb[bty * 8 + (c - 32)];
    else if (c < 44) v = conj_emb[bcj * 4 + (c - 40)];
    else if (c < 48) v = ring_emb[brg * 4 + (c - 44)];
    else if (c < 52) v = stereo_emb[bst * 4 + (c - 48)];
    else if (c < 60) {
      int j = c - 52;
      v = add * refW[j] + dd * refW[8 + j] + has * refW[16 + j] + refb[j];
    } else if (c < 188) v = h[(size_t)src * F + (c - 60)];
    else if (c < 316)   v = h[(size_t)dst * F + (c - 188)];
    else if (c < 444)   v = temb[(size_t)dst * T + (c - 316)];
    else v = 0.f;
    out[c] = (__bf16)v;
  }
}

// ---------------- K2: big GEMM  w = esc(E x 448) @ tpW + tp_b ----------------
// block = 128 threads (4 waves); block per 16-edge M tile; wave covers 13 N tiles.
__global__ __launch_bounds__(128) void k2_big_gemm(const __bf16* esc,
                                                   const __bf16* tpWt,
                                                   const float* tp_b,
                                                   __bf16* wout) {
  int row0 = blockIdx.x * 16;
  int wave = threadIdx.x >> 5;
  int lane = threadIdx.x & 31;
  int hi = (lane >> 4) & 1, col = lane & 15;
  v8f acc[13];
#pragma unroll
  for (int t = 0; t < 13; ++t) {
    float b = tp_b[(wave * 13 + t) * 16 + col];
#pragma unroll
    for (int r = 0; r < 8; ++r) acc[t][r] = b;
  }
  for (int k0 = 0; k0 < KA; k0 += 32) {
    v16bf a = load_a_bf16(esc, KA, row0, k0, lane);
#pragma unroll
    for (int t = 0; t < 13; ++t) {
      v16bf b = load_b_bf16(tpWt, KA, (wave * 13 + t) * 16, k0, lane);
      acc[t] = wmma_bf16(a, b, acc[t]);
    }
  }
#pragma unroll
  for (int t = 0; t < 13; ++t) {
    int n0 = (wave * 13 + t) * 16;
#pragma unroll
    for (int r = 0; r < 8; ++r)
      wout[(size_t)(row0 + hi * 8 + r) * NW + n0 + col] = (__bf16)acc[t][r];
  }
}

// ---------------- K3: message build + scatter (block per edge) ----------------
__global__ __launch_bounds__(128) void k3_msg(const float* h, const int* eidx,
                                              const __bf16* wbf, const float* geom,
                                              const float* mix_s,
                                              const float* mix_1o,
                                              const float* mix_1e,
                                              const float* mix_2, float* msg) {
  __shared__ float P[2496];  // P0[224] | P1[3*256] | P2[3*128] | P3[5*224]
  int e = blockIdx.x;
  int tid = threadIdx.x;
  int src = eidx[e], dst = eidx[N_EDGES + e];
  const float* g = geom + (size_t)e * 8;
  float y1x = g[0], y1y = g[1], y1z = g[2];
  float y2[5] = {g[3], g[4], g[5], g[6], g[7]};
  float y1[3] = {y1x, y1y, y1z};
  const __bf16* we = wbf + (size_t)e * NW;
  const float* hs = h + (size_t)src * F;
  const float INV_SQ3 = 0.57735027f;

  // ---- phase 1: pre-mix tensors ----
  for (int k = tid; k < 224; k += 128) {  // P0
    float v;
    if (k < 128) v = (float)we[k] * hs[k];
    else if (k < 192) {
      int vi = k - 128;
      const float* a = hs + S + vi * 3;
      v = (float)we[128 + vi] * (a[0] * y1x + a[1] * y1y + a[2] * y1z);
    } else {
      int li = k - 192;
      const float* a = hs + S + 6 * V + li * 5;
      v = (float)we[192 + li] *
          (a[0] * y2[0] + a[1] * y2[1] + a[2] * y2[2] + a[3] * y2[3] + a[4] * y2[4]);
    }
    P[k] = v;
  }
  for (int t = tid; t < 768; t += 128) {  // P1 (c,k)
    int c = t / 256, k = t % 256;
    float v;
    if (k < 128) v = (float)we[224 + k] * hs[k] * y1[c];
    else if (k < 192) {
      int vi = k - 128;
      v = (float)we[352 + vi] * hs[S + vi * 3 + c];
    } else {
      int vi = k - 192;
      const float* a = hs + S + 3 * V + vi * 3;  // v1e
      float cx = a[1] * y1z - a[2] * y1y;
      float cy = a[2] * y1x - a[0] * y1z;
      float cz = a[0] * y1y - a[1] * y1x;
      float cr = (c == 0) ? cx : (c == 1 ? cy : cz);
      v = (float)we[416 + vi] * cr;
    }
    P[224 + c * 256 + k] = v;
  }
  for (int t = tid; t < 384; t += 128) {  // P2 (c,k)
    int c = t / 128, k = t % 128;
    float v;
    if (k < 64) v = (float)we[480 + k] * hs[S + 3 * V + k * 3 + c];
    else {
      int vi = k - 64;
      const float* a = hs + S + vi * 3;  // v1o
      float cx = a[1] * y1z - a[2] * y1y;
      float cy = a[2] * y1x - a[0] * y1z;
      float cz = a[0] * y1y - a[1] * y1x;
      float cr = (c == 0) ? cx : (c == 1 ? cy : cz);
      v = (float)we[544 + vi] * cr;
    }
    P[992 + c * 128 + k] = v;
  }
  for (int t = tid; t < 1120; t += 128) {  // P3 (c,k)
    int c = t / 224, k = t % 224;
    float v;
    if (k < 32) v = (float)we[608 + k] * hs[S + 6 * V + k * 5 + c];
    else if (k < 160) {
      int j = k - 32;
      v = (float)we[640 + j] * hs[j] * y2[c];
    } else {
      int vi = k - 160;
      const float* a = hs + S + vi * 3;  // v1o
      float o;
      switch (c) {
        case 0: o = a[0] * y1y + a[1] * y1x; break;
        case 1: o = a[1] * y1z + a[2] * y1y; break;
        case 2: o = (2.f * a[2] * y1z - a[0] * y1x - a[1] * y1y) * INV_SQ3; break;
        case 3: o = a[0] * y1z + a[2] * y1x; break;
        default: o = a[0] * y1x - a[1] * y1y; break;
      }
      v = (float)we[768 + vi] * o;
    }
    P[1376 + c * 224 + k] = v;
  }
  __syncthreads();

  // ---- phase 2: small mixes + atomic scatter ----
  float* md = msg + (size_t)dst * F;
  if (tid < 128) {  // m0 -> s
    float acc = 0.f;
    for (int k = 0; k < 224; ++k) acc += P[k] * mix_s[k * S + tid];
    atomicAdd(&md[tid], DEG_NORM * acc);
  }
  for (int t = tid; t < 192; t += 128) {  // m1 -> v1o
    int vo = t & 63, c = t >> 6;
    const float* p = P + 224 + c * 256;
    float acc = 0.f;
    for (int k = 0; k < 256; ++k) acc += p[k] * mix_1o[k * V + vo];
    atomicAdd(&md[S + vo * 3 + c], DEG_NORM * acc);
  }
  for (int t = tid; t < 192; t += 128) {  // m2 -> v1e
    int vo = t & 63, c = t >> 6;
    const float* p = P + 992 + c * 128;
    float acc = 0.f;
    for (int k = 0; k < 128; ++k) acc += p[k] * mix_1e[k * V + vo];
    atomicAdd(&md[S + 3 * V + vo * 3 + c], DEG_NORM * acc);
  }
  for (int t = tid; t < 160; t += 128) {  // m3 -> l2
    int lo = t & 31, c = t >> 5;
    const float* p = P + 1376 + c * 224;
    float acc = 0.f;
    for (int k = 0; k < 224; ++k) acc += p[k] * mix_2[k * L + lo];
    atomicAdd(&md[S + 6 * V + lo * 5 + c], DEG_NORM * acc);
  }
}

// ---------------- K4a: s = h_s + silu(msg_s @ proj_s) ----------------
__global__ __launch_bounds__(256) void k4_proj_s(const float* msg,
                                                 const __bf16* projT,
                                                 const float* h, float* out) {
  int row0 = blockIdx.x * 16;
  int wave = threadIdx.x >> 5;  // n-tile 0..7
  int lane = threadIdx.x & 31;
  int hi = (lane >> 4) & 1, col = lane & 15;
  v8f acc;
#pragma unroll
  for (int r = 0; r < 8; ++r) acc[r] = 0.f;
  const float* rb = msg + (size_t)(row0 + col) * F;  // A row for this lane
  for (int k0 = 0; k0 < S; k0 += 32) {
    v16bf a = load_a_f32(rb, 1, k0, hi);
    v16bf b = load_b_bf16(projT, 128, wave * 16, k0, lane);
    acc = wmma_bf16(a, b, acc);
  }
#pragma unroll
  for (int r = 0; r < 8; ++r) {
    int row = row0 + hi * 8 + r;
    int c = wave * 16 + col;
    float x = acc[r];
    float sl = x / (1.f + expf(-x));
    out[(size_t)row * F + c] = h[(size_t)row * F + c] + sl;
  }
}

// ---------------- K4b: per-component vector einsum (v1o / v1e) ----------------
// block = 384 threads = 12 waves: wave -> (c = w>>2, ntile = w&3). K=64.
__global__ __launch_bounds__(384) void k4_proj_v(const float* msg,
                                                 const __bf16* projT,
                                                 const float* h, float* out,
                                                 int featOff) {
  int row0 = blockIdx.x * 16;
  int wave = threadIdx.x >> 5;
  int lane = threadIdx.x & 31;
  int c = wave >> 2, nt = wave & 3;
  int hi = (lane >> 4) & 1, col = lane & 15;
  v8f acc;
#pragma unroll
  for (int r = 0; r < 8; ++r) acc[r] = 0.f;
  const float* rb = msg + (size_t)(row0 + col) * F + featOff + c;
  for (int k0 = 0; k0 < V; k0 += 32) {
    v16bf a = load_a_f32(rb, 3, k0, hi);
    v16bf b = load_b_bf16(projT, 64, nt * 16, k0, lane);
    acc = wmma_bf16(a, b, acc);
  }
#pragma unroll
  for (int r = 0; r < 8; ++r) {
    int row = row0 + hi * 8 + r;
    int vo = nt * 16 + col;
    size_t idx = (size_t)row * F + featOff + vo * 3 + c;
    out[idx] = h[idx] + acc[r];
  }
}

// ---------------- K4c: l2 einsum ----------------
// block = 320 threads = 10 waves: wave -> (c = w>>1, ntile = w&1). K=32 (one step).
__global__ __launch_bounds__(320) void k4_proj_l2(const float* msg,
                                                  const __bf16* projT,
                                                  const float* h, float* out) {
  int row0 = blockIdx.x * 16;
  int wave = threadIdx.x >> 5;
  int lane = threadIdx.x & 31;
  int c = wave >> 1, nt = wave & 1;
  int hi = (lane >> 4) & 1, col = lane & 15;
  const int featOff = S + 6 * V;
  v8f acc;
#pragma unroll
  for (int r = 0; r < 8; ++r) acc[r] = 0.f;
  const float* rb = msg + (size_t)(row0 + col) * F + featOff + c;
  v16bf a = load_a_f32(rb, 5, 0, hi);
  v16bf b = load_b_bf16(projT, 32, nt * 16, 0, lane);
  acc = wmma_bf16(a, b, acc);
#pragma unroll
  for (int r = 0; r < 8; ++r) {
    int row = row0 + hi * 8 + r;
    int lo = nt * 16 + col;
    size_t idx = (size_t)row * F + featOff + lo * 5 + c;
    out[idx] = h[idx] + acc[r];
  }
}

// ---------------- K5a: ada = t_emb @ ada_W + ada_b ----------------
// block = 256 threads = 8 waves; each wave does 4 N-tiles (26 total, guarded).
__global__ __launch_bounds__(256) void k5_ada(const float* temb,
                                              const __bf16* adaT,
                                              const float* ada_b, float* ada) {
  int row0 = blockIdx.x * 16;
  int wave = threadIdx.x >> 5;
  int lane = threadIdx.x & 31;
  int hi = (lane >> 4) & 1, col = lane & 15;
  v8f acc[4];
#pragma unroll
  for (int j = 0; j < 4; ++j) {
    int nt = wave * 4 + j;
    float b = (nt < 26) ? ada_b[nt * 16 + col] : 0.f;
#pragma unroll
    for (int r = 0; r < 8; ++r) acc[j][r] = b;
  }
  const float* rb = temb + (size_t)(row0 + col) * T;
  for (int k0 = 0; k0 < T; k0 += 32) {
    v16bf a = load_a_f32(rb, 1, k0, hi);
#pragma unroll
    for (int j = 0; j < 4; ++j) {
      int nt = wave * 4 + j;
      if (nt < 26) {
        v16bf b = load_b_bf16(adaT, 128, nt * 16, k0, lane);
        acc[j] = wmma_bf16(a, b, acc[j]);
      }
    }
  }
#pragma unroll
  for (int j = 0; j < 4; ++j) {
    int nt = wave * 4 + j;
    if (nt < 26) {
#pragma unroll
      for (int r = 0; r < 8; ++r)
        ada[(size_t)(row0 + hi * 8 + r) * 416 + nt * 16 + col] = acc[j][r];
    }
  }
}

// ---------------- K5b: adaLN + vnorms (block per node) ----------------
__device__ __forceinline__ float block_reduce_sum128(float v, volatile float* red,
                                                     int tid) {
  red[tid] = v;
  __syncthreads();
#pragma unroll
  for (int s = 64; s > 0; s >>= 1) {
    if (tid < s) red[tid] += red[tid + s];
    __syncthreads();
  }
  float r = red[0];
  __syncthreads();
  return r;
}

__global__ __launch_bounds__(128) void k5_norm(float* out, const float* ada) {
  __shared__ float red[128];
  int n = blockIdx.x, tid = threadIdx.x;
  float* row = out + (size_t)n * F;
  const float* a = ada + (size_t)n * 416;

  // scalar layernorm + shift/scale
  float x = row[tid];
  float mean = block_reduce_sum128(x, red, tid) * (1.f / 128.f);
  float d = x - mean;
  float var = block_reduce_sum128(d * d, red, tid) * (1.f / 128.f);
  float xn = d * rsqrtf(var + EPS);
  row[tid] = xn * (1.f + a[S + tid]) + a[tid];

  // v1o
  {
    float v0 = row[S + tid], v1 = (tid + 128 < 192) ? row[S + tid + 128] : 0.f;
    float tot = block_reduce_sum128(v0 * v0 + v1 * v1, red, tid);
    float nrm = rsqrtf(tot / (float)V + EPS);
    row[S + tid] = v0 * nrm * (1.f + a[2 * S + tid / 3]);
    if (tid + 128 < 192)
      row[S + tid + 128] = v1 * nrm * (1.f + a[2 * S + (tid + 128) / 3]);
  }
  // v1e
  {
    const int base = S + 3 * V;
    float v0 = row[base + tid], v1 = (tid + 128 < 192) ? row[base + tid + 128] : 0.f;
    float tot = block_reduce_sum128(v0 * v0 + v1 * v1, red, tid);
    float nrm = rsqrtf(tot / (float)V + EPS);
    row[base + tid] = v0 * nrm * (1.f + a[2 * S + V + tid / 3]);
    if (tid + 128 < 192)
      row[base + tid + 128] = v1 * nrm * (1.f + a[2 * S + V + (tid + 128) / 3]);
  }
  // l2
  {
    const int base = S + 6 * V;
    float v0 = (tid < 160) ? row[base + tid] : 0.f;
    float tot = block_reduce_sum128(v0 * v0, red, tid);
    float nrm = rsqrtf(tot / (float)L + EPS);
    if (tid < 160) row[base + tid] = v0 * nrm * (1.f + a[2 * S + 2 * V + tid / 5]);
  }
}

// ---------------- launch ----------------
extern "C" void kernel_launch(void* const* d_in, const int* in_sizes, int n_in,
                              void* d_out, int out_size, void* d_ws, size_t ws_size,
                              hipStream_t stream) {
  (void)in_sizes; (void)n_in; (void)out_size; (void)ws_size;
  const float* h      = (const float*)d_in[0];
  const float* coords = (const float*)d_in[1];
  const int*   eidx   = (const int*)d_in[2];
  const int*   etype  = (const int*)d_in[3];
  const int*   ebt    = (const int*)d_in[4];
  const int*   ebc    = (const int*)d_in[5];
  const int*   ebr    = (const int*)d_in[6];
  const int*   ebs    = (const int*)d_in[7];
  const float* eref   = (const float*)d_in[8];
  const float* temb   = (const float*)d_in[9];
  const float* type_emb   = (const float*)d_in[10];
  const float* bt_emb     = (const float*)d_in[11];
  const float* conj_emb   = (const float*)d_in[12];
  const float* ring_emb   = (const float*)d_in[13];
  const float* stereo_emb = (const float*)d_in[14];
  const float* refW   = (const float*)d_in[15];
  const float* refb   = (const float*)d_in[16];
  const float* tpW    = (const float*)d_in[17];
  const float* tp_b   = (const float*)d_in[18];
  const float* mix_s  = (const float*)d_in[19];
  const float* mix_1o = (const float*)d_in[20];
  const float* mix_1e = (const float*)d_in[21];
  const float* mix_2  = (const float*)d_in[22];
  const float* projs  = (const float*)d_in[23];
  const float* proj1o = (const float*)d_in[24];
  const float* proj1e = (const float*)d_in[25];
  const float* proj2  = (const float*)d_in[26];
  const float* adaW   = (const float*)d_in[27];
  const float* adab   = (const float*)d_in[28];
  float* out = (float*)d_out;

  char* ws = (char*)d_ws;
  __bf16* esc    = (__bf16*)(ws + ESC_OFF);
  float*  geom   = (float*)(ws + GEOM_OFF);
  __bf16* wbf    = (__bf16*)(ws + WBF_OFF);
  float*  msg    = (float*)(ws + MSG_OFF);
  float*  ada    = (float*)(ws + ADA_OFF);
  __bf16* tpWt   = (__bf16*)(ws + TPWT_OFF);
  __bf16* projsT = (__bf16*)(ws + PROJS_OFF);
  __bf16* p1oT   = (__bf16*)(ws + P1O_OFF);
  __bf16* p1eT   = (__bf16*)(ws + P1E_OFF);
  __bf16* p2T    = (__bf16*)(ws + P2T_OFF);
  __bf16* adaT   = (__bf16*)(ws + ADAT_OFF);

  hipMemsetAsync(msg, 0, (size_t)N_NODES * F * sizeof(float), stream);

  k0_prep<<<1024, 256, 0, stream>>>(tpW, projs, proj1o, proj1e, proj2, adaW, tpWt,
                                    projsT, p1oT, p1eT, p2T, adaT);
  k1_edge_prep<<<N_EDGES / 8, 256, 0, stream>>>(
      h, coords, eidx, etype, ebt, ebc, ebr, ebs, eref, temb, type_emb, bt_emb,
      conj_emb, ring_emb, stereo_emb, refW, refb, esc, geom);
  k2_big_gemm<<<N_EDGES / 16, 128, 0, stream>>>(esc, tpWt, tp_b, wbf);
  k3_msg<<<N_EDGES, 128, 0, stream>>>(h, eidx, wbf, geom, mix_s, mix_1o, mix_1e,
                                      mix_2, msg);
  k4_proj_s<<<N_NODES / 16, 256, 0, stream>>>(msg, projsT, h, out);
  k4_proj_v<<<N_NODES / 16, 384, 0, stream>>>(msg, p1oT, h, out, S);
  k4_proj_v<<<N_NODES / 16, 384, 0, stream>>>(msg, p1eT, h, out, S + 3 * V);
  k4_proj_l2<<<N_NODES / 16, 320, 0, stream>>>(msg, p2T, h, out);
  k5_ada<<<N_NODES / 16, 256, 0, stream>>>(temb, adaT, adab, ada);
  k5_norm<<<N_NODES, 128, 0, stream>>>(out, ada);
}
